// CalibrationFreeFP8Linear_30777735643611
// MI455X (gfx1250) — 8-bit matrix operand order repaired and checked by reading, compiled, not run on hardware
//
#include <hip/hip_runtime.h>
#include <hip/hip_bf16.h>

typedef __attribute__((ext_vector_type(16))) int   v16i;
typedef __attribute__((ext_vector_type(8)))  float v8f;

#define M_DIM 16384
#define N_DIM 2048
#define K_DIM 2048
#define FP8_MAX_F 448.0f

#if defined(__has_builtin)
#if __has_builtin(__builtin_amdgcn_cvt_pk_fp8_f32)
#define HAVE_PK_FP8 1
#endif
#endif

static __device__ __forceinline__ unsigned umax_(unsigned a, unsigned b) { return a > b ? a : b; }

__global__ void absmax_bf16_kernel(const uint4* __restrict__ src, long long n_vec,
                                   unsigned* __restrict__ out_bits) {
  __shared__ unsigned sm[256];
  unsigned m = 0;
  long long i = (long long)blockIdx.x * blockDim.x + threadIdx.x;
  const long long stride = (long long)gridDim.x * blockDim.x;
  for (; i < n_vec; i += stride) {
    uint4 v = src[i];
    unsigned a = v.x & 0x7FFF7FFFu, b = v.y & 0x7FFF7FFFu;
    unsigned c = v.z & 0x7FFF7FFFu, d = v.w & 0x7FFF7FFFu;
    m = umax_(m, umax_(a & 0xFFFFu, a >> 16));
    m = umax_(m, umax_(b & 0xFFFFu, b >> 16));
    m = umax_(m, umax_(c & 0xFFFFu, c >> 16));
    m = umax_(m, umax_(d & 0xFFFFu, d >> 16));
  }
  sm[threadIdx.x] = m;
  __syncthreads();
  for (int s = 128; s > 0; s >>= 1) {
    if ((int)threadIdx.x < s) sm[threadIdx.x] = umax_(sm[threadIdx.x], sm[threadIdx.x + s]);
    __syncthreads();
  }
  if (threadIdx.x == 0) atomicMax(out_bits, sm[0]);
}

__global__ void scales_kernel(const unsigned* __restrict__ amax_bits,
                              const float* __restrict__ in_scale,
                              const float* __restrict__ w_scale,
                              float* __restrict__ sc) {
  float ax = __uint_as_float(amax_bits[0] << 16);
  float aw = __uint_as_float(amax_bits[1] << 16);
  float nsx = fminf(fmaxf(FP8_MAX_F / (ax + 1e-12f), 1e-6f), 1e6f);
  float nsw = fminf(fmaxf(FP8_MAX_F / (aw + 1e-12f), 1e-6f), 1e6f);
  float is = 0.9f * in_scale[0] + 0.1f * nsx;
  float ws = 0.9f * w_scale[0] + 0.1f * nsw;
  sc[0] = is;
  sc[1] = ws;
  sc[2] = 1.0f / (is * ws);
}

static __device__ __forceinline__ unsigned cvt_e4m3_sw(float x) {
  unsigned u = __float_as_uint(x);
  unsigned s = (u >> 24) & 0x80u;
  unsigned au = u & 0x7FFFFFFFu;
  if (au >= 0x43E00000u) return s | 0x7Eu;
  unsigned ur = au + 0x80000u;
  int e = (int)(ur >> 23) - 120;
  unsigned mnt = (ur >> 20) & 7u;
  if (e <= 0) return s;
  return s | ((unsigned)e << 3) | mnt;
}

__global__ void quant_kernel(const uint4* __restrict__ src,
                             unsigned long long* __restrict__ dst,
                             const float* __restrict__ sc, int which,
                             long long n_vec) {
  long long i = (long long)blockIdx.x * blockDim.x + threadIdx.x;
  if (i >= n_vec) return;
  const float s = sc[which];
  uint4 v = src[i];
  float f[8];
  f[0] = __uint_as_float((v.x & 0xFFFFu) << 16) * s;
  f[1] = __uint_as_float(v.x & 0xFFFF0000u) * s;
  f[2] = __uint_as_float((v.y & 0xFFFFu) << 16) * s;
  f[3] = __uint_as_float(v.y & 0xFFFF0000u) * s;
  f[4] = __uint_as_float((v.z & 0xFFFFu) << 16) * s;
  f[5] = __uint_as_float(v.z & 0xFFFF0000u) * s;
  f[6] = __uint_as_float((v.w & 0xFFFFu) << 16) * s;
  f[7] = __uint_as_float(v.w & 0xFFFF0000u) * s;
#if defined(HAVE_PK_FP8)
  int lo = 0, hi = 0;
  lo = __builtin_amdgcn_cvt_pk_fp8_f32(f[0], f[1], lo, false);
  lo = __builtin_amdgcn_cvt_pk_fp8_f32(f[2], f[3], lo, true);
  hi = __builtin_amdgcn_cvt_pk_fp8_f32(f[4], f[5], hi, false);
  hi = __builtin_amdgcn_cvt_pk_fp8_f32(f[6], f[7], hi, true);
  unsigned long long packed =
      ((unsigned long long)(unsigned)hi << 32) | (unsigned long long)(unsigned)lo;
#else
  unsigned long long packed = 0;
#pragma unroll
  for (int j = 0; j < 8; ++j)
    packed |= (unsigned long long)cvt_e4m3_sw(f[j]) << (8 * j);
#endif
  dst[i] = packed;
}

__global__ __launch_bounds__(256) void fp8_gemm_kernel(
    const unsigned char* __restrict__ A,
    const unsigned char* __restrict__ W,
    const float* __restrict__ sc,
    __hip_bfloat16* __restrict__ out) {
  const int lane = threadIdx.x & 31;
  const int wid  = threadIdx.x >> 5;
  const int waveM = wid >> 2;
  const int waveN = wid & 3;
  const int mBase = blockIdx.y * 128 + waveM * 64;
  const int nBase = blockIdx.x * 128 + waveN * 32;
  const int lr = lane & 15;
  const int lh = lane >> 4;

  v8f acc[4][2];
#pragma unroll
  for (int i = 0; i < 4; ++i)
#pragma unroll
    for (int j = 0; j < 2; ++j) acc[i][j] = (v8f){};

  const unsigned char* aPtr[4];
#pragma unroll
  for (int i = 0; i < 4; ++i)
    aPtr[i] = A + (long long)(mBase + i * 16 + lr) * K_DIM + lh * 8;
  const unsigned char* bPtr[2];
#pragma unroll
  for (int j = 0; j < 2; ++j)
    bPtr[j] = W + (long long)(nBase + j * 16 + lr) * K_DIM + lh * 8;

  for (int k0 = 0; k0 < K_DIM; k0 += 128) {
    v16i a[4];
#pragma unroll
    for (int i = 0; i < 4; ++i) {
      const unsigned char* p = aPtr[i] + k0;
#pragma unroll
      for (int c = 0; c < 8; ++c) {
        uint2 d = *(const uint2*)(p + c * 16);
        a[i][2 * c + 0] = (int)d.x;
        a[i][2 * c + 1] = (int)d.y;
      }
    }
    v16i b[2];
#pragma unroll
    for (int j = 0; j < 2; ++j) {
      const unsigned char* p = bPtr[j] + k0;
#pragma unroll
      for (int c = 0; c < 8; ++c) {
        uint2 d = *(const uint2*)(p + c * 16);
        b[j][2 * c + 0] = (int)d.x;
        b[j][2 * c + 1] = (int)d.y;
      }
    }
#pragma unroll
    for (int i = 0; i < 4; ++i)
#pragma unroll
      for (int j = 0; j < 2; ++j)
        acc[i][j] = __builtin_amdgcn_wmma_f32_16x16x128_fp8_fp8(
            a[i], b[j], (short)0, acc[i][j], false, false);
  }

  const float inv = sc[2];
#pragma unroll
  for (int i = 0; i < 4; ++i) {
    const int m0 = mBase + i * 16 + lh * 8;
#pragma unroll
    for (int j = 0; j < 2; ++j) {
      const int n = nBase + j * 16 + lr;
#pragma unroll
      for (int r = 0; r < 8; ++r)
        out[(long long)(m0 + r) * N_DIM + n] = __float2bfloat16(acc[i][j][r] * inv);
    }
  }
}

extern "C" void kernel_launch(void* const* d_in, const int* in_sizes, int n_in,
                              void* d_out, int out_size, void* d_ws, size_t ws_size,
                              hipStream_t stream) {
  const uint4* x_bf16 = (const uint4*)d_in[0];
  const uint4* w_bf16 = (const uint4*)d_in[1];
  const float* in_scale = (const float*)d_in[2];
  const float* w_scale  = (const float*)d_in[3];
  __hip_bfloat16* out = (__hip_bfloat16*)d_out;

  char* ws = (char*)d_ws;
  unsigned* amax = (unsigned*)ws;
  float* sc = (float*)(ws + 8);
  unsigned char* w_fp8 = (unsigned char*)(ws + 64);
  unsigned char* x_fp8 = w_fp8 + (size_t)N_DIM * K_DIM;

  hipMemsetAsync(ws, 0, 64, stream);

  const long long nvec_x = (long long)M_DIM * K_DIM / 8;
  const long long nvec_w = (long long)N_DIM * K_DIM / 8;
  absmax_bf16_kernel<<<512, 256, 0, stream>>>(x_bf16, nvec_x, amax + 0);
  absmax_bf16_kernel<<<128, 256, 0, stream>>>(w_bf16, nvec_w, amax + 1);
  scales_kernel<<<1, 1, 0, stream>>>(amax, in_scale, w_scale, sc);

  quant_kernel<<<(unsigned)(nvec_x / 256), 256, 0, stream>>>(
      x_bf16, (unsigned long long*)x_fp8, sc, 0, nvec_x);
  quant_kernel<<<(unsigned)(nvec_w / 256), 256, 0, stream>>>(
      w_bf16, (unsigned long long*)w_fp8, sc, 1, nvec_w);

  dim3 grid(N_DIM / 128, M_DIM / 128);
  fp8_gemm_kernel<<<grid, 256, 0, stream>>>(x_fp8, w_fp8, sc, out);
}
